// BiPixelMambaLayer_46428596469826
// MI455X (gfx1250) — matrix operand order repaired and checked by reading, not compiled, not run on hardware
//
#include <hip/hip_runtime.h>
#include <hip/hip_fp16.h>
#include <cstdint>

#define LSEQ  1024
#define NBTOT 100
#define DM    96
#define DI    192
#define PIX   102400   // 320*320

typedef __attribute__((ext_vector_type(16))) _Float16 v16h;
typedef __attribute__((ext_vector_type(8)))  _Float16 v8h;
typedef __attribute__((ext_vector_type(8)))  float    v8f;
typedef __attribute__((ext_vector_type(4)))  int      v4i;

union H16 { v16h v; v8h h[2]; };

// ---------------------------------------------------------------------------
// Async memory->LDS copy (CDNA5 GLOBAL_LOAD_ASYNC_TO_LDS_B128, ASYNCcnt),
// guarded so the file still compiles if the toolchain lacks the builtins.
// Builtin signature (per hipcc diagnostic): (AS1 int4*, AS3 int4*, Ii, Ii).
// ---------------------------------------------------------------------------
#if __has_builtin(__builtin_amdgcn_global_load_async_to_lds_b128) && \
    __has_builtin(__builtin_amdgcn_s_wait_asynccnt)
#define ASYNC_LDS 1
typedef __attribute__((address_space(1))) v4i as1_v4i;
typedef __attribute__((address_space(3))) v4i as3_v4i;
#endif

__device__ __forceinline__ void async_copy16(const void* g, void* l) {
#ifdef ASYNC_LDS
  __builtin_amdgcn_global_load_async_to_lds_b128((as1_v4i*)g, (as3_v4i*)l, 0, 0);
#else
  *(v8h*)l = *(const v8h*)g;
#endif
}

__device__ __forceinline__ void async_wait0() {
#ifdef ASYNC_LDS
  __builtin_amdgcn_s_wait_asynccnt(0);
#endif
}

// ---------------------------------------------------------------------------
// Kernel 0: swizzle fp32 weights into per-lane WMMA B-fragment layout (f16).
// B[k][n] layout per fragment: entry (lane,j): hi=lane>>4, n=lane&15,
// k = kc*32 + (j>>3)*16 + hi*8 + (j&7).  in_proj: 2 dirs x 24 ntiles x 3 kchunks.
// out_proj: 6 ntiles x 6 kchunks.
// ---------------------------------------------------------------------------
__global__ __launch_bounds__(256)
void swizzle_weights(const float* __restrict__ inWf, const float* __restrict__ inWb,
                     const float* __restrict__ outW,
                     _Float16* __restrict__ wswzIn, _Float16* __restrict__ wswzOut)
{
  int i = blockIdx.x * 256 + threadIdx.x;
  const int IN_TOT = 2 * 24 * 3 * 512;      // 73728
  const int OUT_TOT = 6 * 6 * 512;          // 18432
  if (i < IN_TOT) {
    int dir  = i / (24 * 3 * 512);
    int rem  = i % (24 * 3 * 512);
    int frag = rem / 512;
    int e    = rem % 512;
    int ntile = frag / 3, kc = frag % 3;
    int lane = e / 16, j = e % 16;
    int hi = lane >> 4, n = lane & 15;
    int k = kc * 32 + (j >> 3) * 16 + hi * 8 + (j & 7); // 0..95
    const float* W = dir ? inWb : inWf;                 // [384][96]
    wswzIn[i] = (_Float16)W[(ntile * 16 + n) * DM + k];
  } else if (i < IN_TOT + OUT_TOT) {
    int i2 = i - IN_TOT;
    int frag = i2 / 512;
    int e    = i2 % 512;
    int ntile = frag / 6, kc = frag % 6;
    int lane = e / 16, j = e % 16;
    int hi = lane >> 4, n = lane & 15;
    int k = kc * 32 + (j >> 3) * 16 + hi * 8 + (j & 7); // 0..191
    wswzOut[i2] = (_Float16)outW[(ntile * 16 + n) * DI + k];  // [96][192]
  }
}

// ---------------------------------------------------------------------------
// Kernel 1: patch extraction + LayerNorm -> xn16 [T][96] f16
// ---------------------------------------------------------------------------
__global__ __launch_bounds__(256)
void ln_kernel(const float* __restrict__ x, const float* __restrict__ lnw,
               const float* __restrict__ lnb, _Float16* __restrict__ xn16,
               int nbBase, int T)
{
  int g = blockIdx.x * 256 + threadIdx.x;
  if (g >= T) return;
  int nb = nbBase + (g >> 10);
  int t  = g & 1023;
  int ih = t >> 5, iw = t & 31;
  int ph = nb / 10, pw = nb - ph * 10;
  int base = (ih * 10 + ph) * 320 + (iw * 10 + pw);
  float sum = 0.f, sq = 0.f;
  for (int c = 0; c < DM; ++c) {
    float v = x[(size_t)c * PIX + base];
    sum += v; sq += v * v;
  }
  float mu = sum * (1.f / DM);
  float var = sq * (1.f / DM) - mu * mu;
  float rs = rsqrtf(var + 1e-5f);
  _Float16* row = xn16 + (size_t)g * DM;
  for (int c = 0; c < DM; ++c) {
    float v = x[(size_t)c * PIX + base];
    row[c] = (_Float16)(((v - mu) * rs) * lnw[c] + lnb[c]);
  }
}

// ---------------------------------------------------------------------------
// Kernel 2: in_proj GEMM  xz = xn @ inW.T   (M=T, K=96, N=384) per direction.
// ---------------------------------------------------------------------------
__global__ __launch_bounds__(256)
void inproj_gemm(const _Float16* __restrict__ xn16,
                 const _Float16* __restrict__ wswzIn,
                 _Float16* __restrict__ u16, _Float16* __restrict__ z16, int T)
{
  const int dir  = blockIdx.y;
  const int tid  = threadIdx.x;
  const int wave = tid >> 5, lane = tid & 31;
  const int m = lane & 15, hi = lane >> 4;
  const size_t rowBase = ((size_t)blockIdx.x * 8 + wave) * 16;

  const v8h* rowp = reinterpret_cast<const v8h*>(xn16 + (rowBase + m) * DM);
  v16h A[3];
#pragma unroll
  for (int kc = 0; kc < 3; ++kc) {
    H16 tmp;
    tmp.h[0] = rowp[kc * 4 + hi];       // halves K = kc*32 + hi*8 + 0..7
    tmp.h[1] = rowp[kc * 4 + hi + 2];   // halves K = kc*32 + 16 + hi*8 + 0..7
    A[kc] = tmp.v;
  }

  const _Float16* wb = wswzIn + (size_t)dir * (24 * 3 * 512);
  _Float16* ud = u16 + (size_t)dir * T * DI;
  _Float16* zd = z16 + (size_t)dir * T * DI;

  for (int ntile = 0; ntile < 24; ++ntile) {
    v8f acc = {};
#pragma unroll
    for (int kc = 0; kc < 3; ++kc) {
      v16h B = *reinterpret_cast<const v16h*>(wb + ((size_t)(ntile * 3 + kc) * 512) + lane * 16);
      acc = __builtin_amdgcn_wmma_f32_16x16x32_f16(false, A[kc], false, B,
                                                   (short)0, acc, false, false);
    }
    int col = ntile * 16 + m;
    _Float16* dst; int c;
    if (col < DI) { dst = ud; c = col; } else { dst = zd; c = col - DI; }
    size_t base = (rowBase + (size_t)hi * 8) * DI + c;
#pragma unroll
    for (int r = 0; r < 8; ++r)
      dst[base + (size_t)r * DI] = (_Float16)acc[r];
  }
}

// ---------------------------------------------------------------------------
// Kernel 3: causal depthwise conv + SiLU + x_proj + dt_proj + softplus.
// u tile staged to LDS via async global->LDS b128 (interior) + zero fill
// (out-of-range boundary chunks).
// ---------------------------------------------------------------------------
__global__ __launch_bounds__(256)
void convproj_kernel(const _Float16* __restrict__ u16,
                     const float* __restrict__ cwf, const float* __restrict__ cbf,
                     const float* __restrict__ cwb, const float* __restrict__ cbb,
                     const float* __restrict__ xpwf, const float* __restrict__ xpwb,
                     const float* __restrict__ dtwf, const float* __restrict__ dtwb,
                     const float* __restrict__ dtbf, const float* __restrict__ dtbb,
                     _Float16* __restrict__ ucs16, _Float16* __restrict__ dlt16,
                     float* __restrict__ BCbuf, int T)
{
  const int t0  = blockIdx.x * 32;
  const int seq = blockIdx.y;
  const int dir = blockIdx.z;
  const int tid = threadIdx.x;

  const float* cw  = dir ? cwb  : cwf;
  const float* cb  = dir ? cbb  : cbf;
  const float* xpw = dir ? xpwb : xpwf;
  const float* dtw = dir ? dtwb : dtwf;
  const float* dtb = dir ? dtbb : dtbf;

  const size_t dOff = (size_t)dir * T * DI + (size_t)seq * LSEQ * DI;
  const _Float16* ud = u16   + dOff;
  _Float16* ucsd     = ucs16 + dOff;
  _Float16* dltd     = dlt16 + dOff;
  float* BCd = BCbuf + (size_t)dir * T * 32 + (size_t)seq * LSEQ * 32;

  __shared__ _Float16 su[35 * DI];
  __shared__ float    sucs[32 * DI];
  __shared__ float    sxdbl[32 * 38];

  const int t_s = dir ? t0 : t0 - 3;   // first staged token
  // 35 rows x 24 chunks of 16B each
  for (int i = tid; i < 35 * 24; i += 256) {
    int r = i / 24, ck = i - r * 24;
    int t = t_s + r;
    _Float16* lp = su + r * DI + ck * 8;
    if (t >= 0 && t < LSEQ) {
      async_copy16(ud + (size_t)t * DI + ck * 8, lp);
    } else {
      v8h zz = {};
      *(v8h*)lp = zz;
    }
  }
  async_wait0();
  __syncthreads();

  // conv + bias + SiLU
  for (int i = tid; i < 32 * DI; i += 256) {
    int tl = i / DI, c = i - tl * DI;
    float acc = cb[c];
    if (dir == 0) {
#pragma unroll
      for (int k = 0; k < 4; ++k)
        acc += (float)su[(tl + k) * DI + c] * cw[c * 4 + k];
    } else {
#pragma unroll
      for (int j = 0; j < 4; ++j)
        acc += (float)su[(tl + j) * DI + c] * cw[c * 4 + 3 - j];
    }
    float s = acc / (1.f + __expf(-acc));
    sucs[i] = s;
    ucsd[(size_t)(t0 + tl) * DI + c] = (_Float16)s;
  }
  __syncthreads();

  // x_proj: 38 outputs, K = 192
  for (int i = tid; i < 32 * 38; i += 256) {
    int tl = i / 38, o = i - tl * 38;
    const float* w = xpw + o * DI;
    const float* sr = sucs + tl * DI;
    float acc = 0.f;
    for (int c = 0; c < DI; ++c) acc += sr[c] * w[c];
    sxdbl[i] = acc;
    if (o >= 6 && o < 22)      BCd[(size_t)(t0 + tl) * 32 + (o - 6)]       = acc;
    else if (o >= 22)          BCd[(size_t)(t0 + tl) * 32 + 16 + (o - 22)] = acc;
  }
  __syncthreads();

  // dt_proj + softplus: 192 outputs, K = 6
  for (int i = tid; i < 32 * DI; i += 256) {
    int tl = i / DI, d = i - tl * DI;
    float acc = dtb[d];
#pragma unroll
    for (int r = 0; r < 6; ++r) acc += sxdbl[tl * 38 + r] * dtw[d * 6 + r];
    float sp = (acc > 20.f) ? acc : log1pf(__expf(acc));
    dltd[(size_t)(t0 + tl) * DI + d] = (_Float16)sp;
  }
}

// ---------------------------------------------------------------------------
// Kernel 4: selective scan. One block per (seq, dir); thread d owns channel d
// with 16 f32 states in registers. B/C staged in LDS 64 steps at a time via
// async global->LDS copies; next-tile delta/u rows prefetched.
// ---------------------------------------------------------------------------
__global__ __launch_bounds__(192)
void scan_kernel(const _Float16* __restrict__ dlt16, const _Float16* __restrict__ ucs16,
                 const _Float16* __restrict__ z16, const float* __restrict__ BCbuf,
                 _Float16* __restrict__ Y16,
                 const float* __restrict__ Alogf, const float* __restrict__ Alogb,
                 const float* __restrict__ Df, const float* __restrict__ Db, int T)
{
  const int seq = blockIdx.x;
  const int dir = blockIdx.y;
  const int d   = threadIdx.x;        // 0..191

  const size_t off = (size_t)dir * T * DI + (size_t)seq * LSEQ * DI;
  const _Float16* dl = dlt16 + off;
  const _Float16* uc = ucs16 + off;
  const _Float16* zz = z16   + off;
  _Float16* yo       = Y16   + off;
  const float* bc = BCbuf + (size_t)dir * T * 32 + (size_t)seq * LSEQ * 32;

  const float* Alog = dir ? Alogb : Alogf;
  float A[16], s[16];
#pragma unroll
  for (int n = 0; n < 16; ++n) {
    A[n] = -__expf(Alog[d * 16 + n]);
    s[n] = 0.f;
  }
  const float Dp = (dir ? Db : Df)[d];

  __shared__ float sBC[64 * 32];

  for (int tt = 0; tt < 16; ++tt) {
    const int tile = dir ? (15 - tt) : tt;
    __syncthreads();
    // 512 chunks of 16B (2048 floats), always in-range -> pure async path
    for (int i = d; i < 512; i += 192)
      async_copy16(bc + (size_t)tile * 2048 + (size_t)i * 4, sBC + i * 4);
    async_wait0();
    __syncthreads();

    const int nextTile = dir ? tile - 1 : tile + 1;
    if (nextTile >= 0 && nextTile < 16) {
      __builtin_prefetch((const void*)(dl + (size_t)(nextTile * 64) * DI), 0, 1);
      __builtin_prefetch((const void*)(uc + (size_t)(nextTile * 64) * DI), 0, 1);
    }

    for (int ss = 0; ss < 64; ++ss) {
      const int tl = dir ? (63 - ss) : ss;
      const int t  = tile * 64 + tl;
      const float dt  = (float)dl[(size_t)t * DI + d];
      const float ucv = (float)uc[(size_t)t * DI + d];
      const float zv  = (float)zz[(size_t)t * DI + d];
      const float du  = dt * ucv;
      const float* Bp = &sBC[tl * 32];
      const float* Cp = Bp + 16;
      float y = 0.f;
#pragma unroll
      for (int n = 0; n < 16; ++n) {
        s[n] = s[n] * __expf(A[n] * dt) + du * Bp[n];
        y += s[n] * Cp[n];
      }
      y += ucv * Dp;
      const float sz = zv / (1.f + __expf(-zv));
      yo[(size_t)t * DI + d] = (_Float16)(y * sz);
    }
  }
}

// ---------------------------------------------------------------------------
// Kernel 5: out_proj GEMM  y = (Yf + Yb) @ outW.T  (M=T, K=192, N=96),
// fused un-patch scatter + residual add into d_out.
// ---------------------------------------------------------------------------
__global__ __launch_bounds__(256)
void outproj_gemm(const _Float16* __restrict__ Y16, const _Float16* __restrict__ wswzOut,
                  const float* __restrict__ x, float* __restrict__ out,
                  int nbBase, int T)
{
  const int tid  = threadIdx.x;
  const int wave = tid >> 5, lane = tid & 31;
  const int m = lane & 15, hi = lane >> 4;
  const size_t rowBase = ((size_t)blockIdx.x * 8 + wave) * 16;

  const v8h* rf = reinterpret_cast<const v8h*>(Y16 + (rowBase + m) * DI);
  const v8h* rb = reinterpret_cast<const v8h*>(Y16 + (size_t)T * DI + (rowBase + m) * DI);
  v16h A[6];
#pragma unroll
  for (int kc = 0; kc < 6; ++kc) {
    H16 tmp;
    tmp.h[0] = rf[kc * 4 + hi]     + rb[kc * 4 + hi];
    tmp.h[1] = rf[kc * 4 + hi + 2] + rb[kc * 4 + hi + 2];
    A[kc] = tmp.v;
  }

  for (int ntile = 0; ntile < 6; ++ntile) {
    v8f acc = {};
#pragma unroll
    for (int kc = 0; kc < 6; ++kc) {
      v16h B = *reinterpret_cast<const v16h*>(wswzOut + ((size_t)(ntile * 6 + kc) * 512) + lane * 16);
      acc = __builtin_amdgcn_wmma_f32_16x16x32_f16(false, A[kc], false, B,
                                                   (short)0, acc, false, false);
    }
    const int c = ntile * 16 + m;     // output channel 0..95
#pragma unroll
    for (int r = 0; r < 8; ++r) {
      int g  = (int)rowBase + hi * 8 + r;       // chunk-local token
      int G  = nbBase * LSEQ + g;               // global token
      int nb = G >> 10;
      int t  = G & 1023;
      int ih = t >> 5, iw = t & 31;
      int ph = nb / 10, pw = nb - ph * 10;
      size_t idx = (size_t)c * PIX + (ih * 10 + ph) * 320 + (iw * 10 + pw);
      out[idx] = x[idx] + acc[r];
    }
  }
}

// ---------------------------------------------------------------------------
extern "C" void kernel_launch(void* const* d_in, const int* in_sizes, int n_in,
                              void* d_out, int out_size, void* d_ws, size_t ws_size,
                              hipStream_t stream)
{
  (void)in_sizes; (void)n_in; (void)out_size;
  const float* x      = (const float*)d_in[0];
  const float* ln_w   = (const float*)d_in[1];
  const float* ln_b   = (const float*)d_in[2];
  const float* inW_f  = (const float*)d_in[3];
  const float* cw_f   = (const float*)d_in[4];
  const float* cb_f   = (const float*)d_in[5];
  const float* xpW_f  = (const float*)d_in[6];
  const float* dtW_f  = (const float*)d_in[7];
  const float* dtB_f  = (const float*)d_in[8];
  const float* Alog_f = (const float*)d_in[9];
  const float* D_f    = (const float*)d_in[10];
  const float* inW_b  = (const float*)d_in[11];
  const float* cw_b   = (const float*)d_in[12];
  const float* cb_b   = (const float*)d_in[13];
  const float* xpW_b  = (const float*)d_in[14];
  const float* dtW_b  = (const float*)d_in[15];
  const float* dtB_b  = (const float*)d_in[16];
  const float* Alog_b = (const float*)d_in[17];
  const float* D_b    = (const float*)d_in[18];
  const float* outW   = (const float*)d_in[19];
  float* out = (float*)d_out;

  char* ws = (char*)d_ws;
  size_t off = 0;
  auto take = [&](size_t bytes) -> char* {
    char* p = ws + off;
    off = (off + bytes + 255) & ~(size_t)255;
    return p;
  };

  _Float16* wswzIn  = (_Float16*)take((size_t)2 * 24 * 3 * 512 * sizeof(_Float16));
  _Float16* wswzOut = (_Float16*)take((size_t)6 * 6 * 512 * sizeof(_Float16));
  const size_t fixedOff = off;

  // 4288 bytes/token * 1024 tokens/seq
  const size_t perSeq = (size_t)4288 * LSEQ;
  size_t avail = (ws_size > fixedOff + 4096) ? (ws_size - fixedOff - 4096) : 0;
  int CH = (int)(avail / perSeq);
  if (CH < 1) CH = 1;
  if (CH > NBTOT) CH = NBTOT;
  const int Tmax = CH * LSEQ;

  _Float16* xn16  = (_Float16*)take((size_t)Tmax * DM * 2);
  _Float16* u16   = (_Float16*)take((size_t)2 * Tmax * DI * 2);
  _Float16* z16   = (_Float16*)take((size_t)2 * Tmax * DI * 2);
  _Float16* ucs16 = (_Float16*)take((size_t)2 * Tmax * DI * 2);
  _Float16* dlt16 = (_Float16*)take((size_t)2 * Tmax * DI * 2);
  float*    BCbuf = (float*)   take((size_t)2 * Tmax * 32 * 4);
  _Float16* Y16   = (_Float16*)take((size_t)2 * Tmax * DI * 2);

  {
    const int total = 2 * 24 * 3 * 512 + 6 * 6 * 512;   // 92160
    swizzle_weights<<<dim3((total + 255) / 256), dim3(256), 0, stream>>>(
        inW_f, inW_b, outW, wswzIn, wswzOut);
  }

  for (int nbBase = 0; nbBase < NBTOT; nbBase += CH) {
    int cs = NBTOT - nbBase; if (cs > CH) cs = CH;
    const int T = cs * LSEQ;

    ln_kernel<<<dim3(T / 256), dim3(256), 0, stream>>>(x, ln_w, ln_b, xn16, nbBase, T);

    inproj_gemm<<<dim3(T / 128, 2), dim3(256), 0, stream>>>(xn16, wswzIn, u16, z16, T);

    convproj_kernel<<<dim3(32, cs, 2), dim3(256), 0, stream>>>(
        u16, cw_f, cb_f, cw_b, cb_b, xpW_f, xpW_b, dtW_f, dtW_b, dtB_f, dtB_b,
        ucs16, dlt16, BCbuf, T);

    scan_kernel<<<dim3(cs, 2), dim3(192), 0, stream>>>(
        dlt16, ucs16, z16, BCbuf, Y16, Alog_f, Alog_b, D_f, D_b, T);

    outproj_gemm<<<dim3(T / 128), dim3(256), 0, stream>>>(Y16, wswzOut, x, out, nbBase, T);
  }
}
